// RITSModel_68616397521548
// MI455X (gfx1250) — matrix operand order repaired and checked by reading, not compiled, not run on hardware
//
#include <hip/hip_runtime.h>
#include <hip/hip_bf16.h>

#define T_LEN   252
#define FEAT    21
#define HID     128
#define BATCH   4096
#define ROWS    16            // batch rows per block (one WMMA M-tile)
#define NTHREADS 128          // 4 wave32s

#if defined(__has_include)
#if __has_include(<hip/amd_detail/amd_gfx1250_TDM.h>)
#define RITS_TDM6 1           // amdgpu-toolchain: 6-arg tensor builtin
#endif
#endif

#if __has_builtin(__builtin_amdgcn_tensor_load_to_lds)
#define RITS_HAVE_TDM 1
#else
#define RITS_HAVE_TDM 0
#endif

typedef _Float16 v16h __attribute__((ext_vector_type(16)));
typedef float    v8f  __attribute__((ext_vector_type(8)));
typedef unsigned int u32x4 __attribute__((ext_vector_type(4)));
typedef int          i32x4 __attribute__((ext_vector_type(4)));
typedef int          i32x8 __attribute__((ext_vector_type(8)));

union HPack { unsigned int u[8]; v16h v; };

__device__ __forceinline__ v8f wmma_f16(v16h a, v16h b, v8f c) {
  // D = A(16x32 f16) * B(32x16 f16) + C(16x16 f32)
  return __builtin_amdgcn_wmma_f32_16x16x32_f16(false, a, false, b, (short)0, c,
                                                false, false);
}

__device__ __forceinline__ v8f v8f_zero() {
  v8f z = {0.f, 0.f, 0.f, 0.f, 0.f, 0.f, 0.f, 0.f};
  return z;
}

__device__ __forceinline__ float sigm(float x) { return 1.0f / (1.0f + __expf(-x)); }

__device__ __forceinline__ unsigned int pack2(float a, float b) {
  union { unsigned int u; _Float16 h[2]; } t;
  t.h[0] = (_Float16)a; t.h[1] = (_Float16)b;
  return t.u;
}

// ---- CDNA5 async global->LDS (ASYNCcnt) ------------------------------------
// Flat LDS pointers carry the LDS byte offset in addr[31:0] (ISA 10.2).
__device__ __forceinline__ unsigned int lds_off(const void* p) {
  return (unsigned int)(unsigned long long)(uintptr_t)p;
}
__device__ __forceinline__ void async_b32(unsigned int dst_lds, const void* gptr) {
  asm volatile("global_load_async_to_lds_b32 %0, %1, off"
               :: "v"(dst_lds), "v"((unsigned long long)(uintptr_t)gptr)
               : "memory");
}
__device__ __forceinline__ void wait_async0() {
  asm volatile("s_wait_asynccnt 0x0" ::: "memory");
}

// ---- CDNA5 Tensor Data Mover: 2D tile load (TENSORcnt) ---------------------
// D# per ISA 8.3/8.4: tile = ROWS x FEAT f32, row stride T_LEN*FEAT elements.
__device__ __forceinline__ void tdm_load_tile(unsigned int lds_byte,
                                              const void* gaddr) {
#if RITS_HAVE_TDM
  const unsigned long long ga = (unsigned long long)(uintptr_t)gaddr;
  u32x4 g0;
  g0.x = 1u;                                            // count=1, user, no gather
  g0.y = lds_byte;                                      // lds_addr (bytes)
  g0.z = (unsigned int)ga;                              // global_addr[31:0]
  g0.w = (unsigned int)((ga >> 32) & 0x01FFFFFFull)     // global_addr[56:32]
         | (2u << 30);                                  // type=2 ("image")
  i32x8 g1 = {
    (int)(2u << 16),            // wg_mask=0 | data_size=2 (4B) | no flags
    (int)(FEAT << 16),          // tensor_dim0 = 21 (bits 79:48, low half)
    (int)(ROWS << 16),          // tensor_dim1 = 16 (bits 111:80, low half)
    (int)(FEAT << 16),          // tile_dim0 = 21 (bits 127:112)
    ROWS,                       // tile_dim1 = 16 (bits 143:128); tile_dim2=0
    T_LEN * FEAT,               // tensor_dim0_stride = 5292 (bits 207:160)
    0, 0                        // stride hi / tensor_dim1_stride = 0
  };
  i32x4 g2 = {1, 1, 0, 0};      // tensor_dim2/3 = 1; tile_dim3 = 0 (unused)
  i32x4 g3 = {0, 1 << 16, 0, 0};// tensor_dim4 = 1; tile_dim4 = 0 (unused)
#if defined(RITS_TDM6)
  i32x8 g4 = {0, 0, 0, 0, 0, 0, 0, 0};
  __builtin_amdgcn_tensor_load_to_lds(g0, g1, g2, g3, g4, 0);
#else
  __builtin_amdgcn_tensor_load_to_lds(g0, g1, g2, g3, 0);
#endif
#else
  // fallback: per-lane async DMA of the whole tile (issued by one wave)
  for (int j = 0; j < ROWS * FEAT; ++j) {
    // note: only reachable if TDM builtin is missing; keep it simple
    async_b32(lds_byte + 4u * j,
              (const char*)gaddr + 4ull * ((j / FEAT) * (T_LEN * FEAT) + (j % FEAT)));
  }
#endif
}
__device__ __forceinline__ void wait_tensor0() {
#if RITS_HAVE_TDM
  __builtin_amdgcn_s_wait_tensorcnt(0);
#endif
}

// A-fragment (16x32 f16) from a row-major LDS tile [16][stride].
// Layout: lanes 0-15 -> M=lane, K {0..15};
// lanes 16-31 -> M=lane-16, K {16..31} (contiguous, matches B fragments).
__device__ __forceinline__ v16h load_a(const _Float16* base, int stride, int kbase,
                                       int lane) {
  const int M  = lane & 15;
  const int ko = (lane & 16) ? 16 : 0;
  HPack p;
#pragma unroll
  for (int j = 0; j < 8; ++j) {
    const int k = kbase + ko + 2 * j;
    p.u[j] = *(const unsigned int*)(base + M * stride + k);
  }
  return p.v;
}

// ---------------------------------------------------------------------------
// Pre-pass: per-timestep loss denominators (input-only, decouples blocks).
// ws[t]   = 1/(sum(masks[:,t,:]) + 1e-5)   t in [0,252)
// ws[252] = 1/(sum(is_train) + 1e-5)
// ---------------------------------------------------------------------------
__global__ void rits_denoms(const float* __restrict__ masks,
                            const float* __restrict__ istrain,
                            float* __restrict__ ws) {
  __shared__ float red[256];
  const int tid = threadIdx.x;
  const int t   = blockIdx.x;
  float s = 0.f;
  if (t < T_LEN) {
    for (int i = tid; i < BATCH * FEAT; i += 256) {
      const int b = i / FEAT, f = i % FEAT;
      s += masks[((long)b * T_LEN + t) * FEAT + f];
    }
  } else {
    for (int i = tid; i < BATCH; i += 256) s += istrain[i];
  }
  red[tid] = s;
  __syncthreads();
  for (int off = 128; off > 0; off >>= 1) {
    if (tid < off) red[tid] += red[tid + off];
    __syncthreads();
  }
  if (tid == 0) ws[(t < T_LEN) ? t : T_LEN] = 1.0f / (red[0] + 1e-5f);
}

// ---------------------------------------------------------------------------
// Main recurrent kernel: one block = 16 batch rows for all 252 steps.
// Weights live in per-wave register fragments (~400 VGPRs, CDNA5 1024/wave);
// inputs stream in via double-buffered TDM (values, masks) + async DMA (deltas).
// ---------------------------------------------------------------------------
__global__ void __launch_bounds__(NTHREADS, 1)
rits_main(const float* __restrict__ values, const float* __restrict__ masks,
          const float* __restrict__ deltas, const float* __restrict__ labels,
          const float* __restrict__ istrain,
          const float* __restrict__ Wd,  const float* __restrict__ bd,
          const float* __restrict__ Wr,  const float* __restrict__ br,
          const float* __restrict__ Wih, const float* __restrict__ Whh,
          const float* __restrict__ bih, const float* __restrict__ bhh,
          const float* __restrict__ Wout, const float* __restrict__ bout,
          float* __restrict__ ws, float* __restrict__ out) {
  __shared__ __attribute__((aligned(16))) _Float16 hA[ROWS * 132];   // h f16 (A for x_h & gates)
  __shared__ __attribute__((aligned(16))) _Float16 inpA[ROWS * 68];  // [x_c | m | 0-pad]
  __shared__ __attribute__((aligned(16))) _Float16 dA[ROWS * 36];    // deltas, 0-padded K
  __shared__ __attribute__((aligned(16))) float stg[2][3][ROWS * FEAT]; // DMA staging x/m/d
  __shared__ float xb[ROWS * 22];
  __shared__ float mb[ROWS * 22];
  __shared__ float hf[ROWS * HID];
  __shared__ float red[NTHREADS];

  const int tid   = threadIdx.x;
  const int w     = tid >> 5;        // wave id, 0..3 (wave32)
  const int lane  = tid & 31;
  const int n16   = lane & 15;       // C/D column within tile
  const int mOff  = (lane & 16) ? 8 : 0;  // C/D row offset
  const int kB16  = (lane & 16) ? 16 : 0; // B-fragment K offset for hi lanes
  const int b0    = blockIdx.x * ROWS;

  // ------ loop-invariant B fragments in registers (weights stay resident) ---
  v16h whhF[8][4]; v16h wihF[8][2]; v16h wdF[2]; v16h wrF[4];
  float bg[8]; float bdec[2]; float bregv = 0.f;

#pragma unroll
  for (int t8 = 0; t8 < 8; ++t8) {            // tile t8 = q*2+s (q=gate, s=half)
    const int q = t8 >> 1, s = t8 & 1;
    const int gcol = q * HID + w * 32 + s * 16 + n16;   // gate column (N)
    bg[t8] = bih[gcol] + bhh[gcol];
#pragma unroll
    for (int c = 0; c < 4; ++c) {             // K chunks of 32 (HID=128)
      HPack p;
#pragma unroll
      for (int j = 0; j < 8; ++j) {
        const int k = 32 * c + 2 * j + kB16;
        p.u[j] = pack2(Whh[gcol * HID + k], Whh[gcol * HID + k + 1]);
      }
      whhF[t8][c] = p.v;
    }
#pragma unroll
    for (int c = 0; c < 2; ++c) {             // K chunks (2*FEAT=42, pad 64)
      HPack p;
#pragma unroll
      for (int j = 0; j < 8; ++j) {
        const int k = 32 * c + 2 * j + kB16;
        const float e0 = (k < 2 * FEAT) ? Wih[gcol * (2 * FEAT) + k] : 0.f;
        const float e1 = (k < 2 * FEAT) ? Wih[gcol * (2 * FEAT) + k + 1] : 0.f;
        p.u[j] = pack2(e0, e1);
      }
      wihF[t8][c] = p.v;
    }
  }
#pragma unroll
  for (int s = 0; s < 2; ++s) {               // decay B: [K=21pad32][N=128]
    const int hcol = w * 32 + s * 16 + n16;
    bdec[s] = bd[hcol];
    HPack p;
#pragma unroll
    for (int j = 0; j < 8; ++j) {
      const int k = 2 * j + kB16;
      const float e0 = (k < FEAT)     ? Wd[hcol * FEAT + k]     : 0.f;
      const float e1 = (k + 1 < FEAT) ? Wd[hcol * FEAT + k + 1] : 0.f;
      p.u[j] = pack2(e0, e1);
    }
    wdF[s] = p.v;
  }
  if (w < 2) {                                // regression B: [K=128][N=21pad32]
    const int n = w * 16 + n16;
    bregv = (n < FEAT) ? br[n] : 0.f;
#pragma unroll
    for (int c = 0; c < 4; ++c) {
      HPack p;
#pragma unroll
      for (int j = 0; j < 8; ++j) {
        const int k = 32 * c + 2 * j + kB16;
        const float e0 = (n < FEAT) ? Wr[n * HID + k]     : 0.f;
        const float e1 = (n < FEAT) ? Wr[n * HID + k + 1] : 0.f;
        p.u[j] = pack2(e0, e1);
      }
      wrF[c] = p.v;
    }
  } else {
#pragma unroll
    for (int c = 0; c < 4; ++c) {
      HPack p;
#pragma unroll
      for (int j = 0; j < 8; ++j) p.u[j] = 0u;
      wrF[c] = p.v;
    }
  }

  // zero constant pad regions of the LDS A tiles (done once)
  for (int j = tid; j < ROWS * 36; j += NTHREADS)
    if ((j % 36) >= FEAT) dA[j] = (_Float16)0.f;
  for (int j = tid; j < ROWS * 68; j += NTHREADS)
    if ((j % 68) >= 2 * FEAT) inpA[j] = (_Float16)0.f;

  // kick off DMA of step-0 inputs: TDM for values/masks, async for deltas
  {
    const long base = (long)b0 * (T_LEN * FEAT);
    if (w == 0) tdm_load_tile(lds_off(&stg[0][0][0]), values + base);
    if (w == 1) tdm_load_tile(lds_off(&stg[0][1][0]), masks + base);
    for (int j = tid; j < ROWS * FEAT; j += NTHREADS) {
      const int row = j / FEAT, f = j % FEAT;
      async_b32(lds_off(&stg[0][2][j]),
                deltas + base + (long)row * (T_LEN * FEAT) + f);
    }
  }

  float h_reg[2][8], c_reg[2][8];
#pragma unroll
  for (int s = 0; s < 2; ++s)
#pragma unroll
    for (int r = 0; r < 8; ++r) { h_reg[s][r] = 0.f; c_reg[s][r] = 0.f; }
  float lossAcc = 0.f;
  const float invDenTrain = ws[T_LEN];

#pragma unroll 1
  for (int t = 0; t < T_LEN; ++t) {
    // ---- stage 0: wait DMA, convert staged inputs, issue DMA for t+1 -------
    wait_tensor0();
    wait_async0();
    __syncthreads();                      // staging[t&1] visible to all waves
    {
      const float* sx = stg[t & 1][0];
      const float* sm = stg[t & 1][1];
      const float* sd = stg[t & 1][2];
      for (int j = tid; j < ROWS * FEAT; j += NTHREADS) {
        const int row = j / FEAT, f = j % FEAT;
        const float xv = sx[j];
        const float mv = sm[j];
        const float dv = sd[j];
        xb[row * 22 + f]          = xv;
        mb[row * 22 + f]          = mv;
        dA[row * 36 + f]          = (_Float16)dv;
        inpA[row * 68 + FEAT + f] = (_Float16)mv;
      }
    }
    if (t + 1 < T_LEN) {                  // overlap DMA with stages 1-3
      const int bn = (t + 1) & 1;
      const long base = (long)b0 * (T_LEN * FEAT) + (long)(t + 1) * FEAT;
      if (w == 0) tdm_load_tile(lds_off(&stg[bn][0][0]), values + base);
      if (w == 1) tdm_load_tile(lds_off(&stg[bn][1][0]), masks + base);
      for (int j = tid; j < ROWS * FEAT; j += NTHREADS) {
        const int row = j / FEAT, f = j % FEAT;
        async_b32(lds_off(&stg[bn][2][j]),
                  deltas + base + (long)row * (T_LEN * FEAT) + f);
      }
    }
    if (t + 2 < T_LEN && tid < ROWS) {    // warm L2 two steps ahead
      const long nidx = ((long)(b0 + tid) * T_LEN + (t + 2)) * FEAT;
      __builtin_prefetch(values + nidx, 0, 3);
      __builtin_prefetch(masks  + nidx, 0, 3);
      __builtin_prefetch(deltas + nidx, 0, 3);
    }
    __syncthreads();

    // ---- stage 1: gamma = exp(-relu(d@Wd^T+bd)); h *= gamma; publish h f16 --
    {
      const v16h aD = load_a(dA, 36, 0, lane);
#pragma unroll
      for (int s = 0; s < 2; ++s) {
        v8f acc = v8f_zero();
        acc = wmma_f16(aD, wdF[s], acc);
        const int col = w * 32 + s * 16 + n16;
#pragma unroll
        for (int r = 0; r < 8; ++r) {
          const float g     = acc[r] + bdec[s];
          const float gamma = __expf(-fmaxf(g, 0.f));
          const float hv    = h_reg[s][r] * gamma;
          h_reg[s][r]       = hv;
          hA[(r + mOff) * 132 + col] = (_Float16)hv;
        }
      }
    }
    __syncthreads();

    // ---- stage 2 (waves 0,1): x_h = h@Wr^T+br; x_c; imputation; loss -------
    if (w < 2) {
      v8f acc = v8f_zero();
#pragma unroll
      for (int c = 0; c < 4; ++c)
        acc = wmma_f16(load_a(hA, 132, 32 * c, lane), wrF[c], acc);
      const int n = w * 16 + n16;
      if (n < FEAT) {
        const float invDen = ws[t];
#pragma unroll
        for (int r = 0; r < 8; ++r) {
          const int   row = r + mOff;
          const float xh  = acc[r] + bregv;
          const float xv  = xb[row * 22 + n];
          const float mv  = mb[row * 22 + n];
          const float xc  = mv * xv + (1.f - mv) * xh;
          inpA[row * 68 + n] = (_Float16)xc;
          out[1 + BATCH + (((long)(b0 + row) * T_LEN + t) * FEAT + n)] = xc;
          lossAcc += fabsf(xv - xh) * mv * invDen;
        }
      }
    }
    __syncthreads();

    // ---- stage 3: gates = [x_c|m]@Wih^T + h@Whh^T + b; LSTM update ---------
    {
      const v16h aI0 = load_a(inpA, 68, 0,  lane);
      const v16h aI1 = load_a(inpA, 68, 32, lane);
      const v16h aH0 = load_a(hA, 132, 0,  lane);
      const v16h aH1 = load_a(hA, 132, 32, lane);
      const v16h aH2 = load_a(hA, 132, 64, lane);
      const v16h aH3 = load_a(hA, 132, 96, lane);
#pragma unroll
      for (int s = 0; s < 2; ++s) {
        v8f acc[4];
#pragma unroll
        for (int q = 0; q < 4; ++q) {
          const int t8 = q * 2 + s;
          v8f a = v8f_zero();
          a = wmma_f16(aI0, wihF[t8][0], a);
          a = wmma_f16(aI1, wihF[t8][1], a);
          a = wmma_f16(aH0, whhF[t8][0], a);
          a = wmma_f16(aH1, whhF[t8][1], a);
          a = wmma_f16(aH2, whhF[t8][2], a);
          a = wmma_f16(aH3, whhF[t8][3], a);
          acc[q] = a;
        }
#pragma unroll
        for (int r = 0; r < 8; ++r) {
          const float iv = sigm(acc[0][r] + bg[0 * 2 + s]);
          const float fv = sigm(acc[1][r] + bg[1 * 2 + s]);
          const float gv = tanhf(acc[2][r] + bg[2 * 2 + s]);
          const float ov = sigm(acc[3][r] + bg[3 * 2 + s]);
          const float cc = fv * c_reg[s][r] + iv * gv;
          c_reg[s][r] = cc;
          h_reg[s][r] = ov * tanhf(cc);
        }
      }
    }
    // next iteration's loop-top wait+barrier protects inpA/hA/staging reuse
  }

  // ---- epilogue: y_h, y_loss, block loss partial ----------------------------
#pragma unroll
  for (int s = 0; s < 2; ++s) {
    const int col = w * 32 + s * 16 + n16;
#pragma unroll
    for (int r = 0; r < 8; ++r) hf[(r + mOff) * HID + col] = h_reg[s][r];
  }
  __syncthreads();
  float extra = 0.f;
  if (tid < ROWS) {
    float y = bout[0];
#pragma unroll 4
    for (int k = 0; k < HID; ++k) y += hf[tid * HID + k] * Wout[k];
    out[1 + b0 + tid] = y;
    const float dy = y - labels[b0 + tid];
    extra = dy * dy * istrain[b0 + tid] * invDenTrain;
  }
  red[tid] = lossAcc + extra;
  __syncthreads();
  if (tid == 0) {
    float s = 0.f;
    for (int i = 0; i < NTHREADS; ++i) s += red[i];
    ws[256 + blockIdx.x] = s;   // deterministic fixed-order partial
  }
}

// Fixed-order sum of the 256 block partials -> loss (deterministic, no atomics)
__global__ void rits_finalize(const float* __restrict__ ws, float* __restrict__ out) {
  float s = 0.f;
  for (int i = 0; i < 256; ++i) s += ws[256 + i];
  out[0] = s;
}

extern "C" void kernel_launch(void* const* d_in, const int* in_sizes, int n_in,
                              void* d_out, int out_size, void* d_ws, size_t ws_size,
                              hipStream_t stream) {
  const float* values  = (const float*)d_in[0];
  const float* masks   = (const float*)d_in[1];
  const float* deltas  = (const float*)d_in[2];
  // d_in[3] = evals, d_in[4] = eval_masks : unused by the reference math
  const float* labels  = (const float*)d_in[5];
  const float* istrain = (const float*)d_in[6];
  const float* Wd   = (const float*)d_in[7];
  const float* bd   = (const float*)d_in[8];
  const float* Wr   = (const float*)d_in[9];
  const float* br   = (const float*)d_in[10];
  const float* Wih  = (const float*)d_in[11];
  const float* Whh  = (const float*)d_in[12];
  const float* bih  = (const float*)d_in[13];
  const float* bhh  = (const float*)d_in[14];
  const float* Wout = (const float*)d_in[15];
  const float* bout = (const float*)d_in[16];
  float* out = (float*)d_out;
  float* ws  = (float*)d_ws;

  rits_denoms<<<T_LEN + 1, 256, 0, stream>>>(masks, istrain, ws);
  rits_main<<<BATCH / ROWS, NTHREADS, 0, stream>>>(
      values, masks, deltas, labels, istrain,
      Wd, bd, Wr, br, Wih, Whh, bih, bhh, Wout, bout, ws, out);
  rits_finalize<<<1, 1, 0, stream>>>(ws, out);
}
